// EquiPocket_91328184582717
// MI455X (gfx1250) — matrix operand order repaired and checked by reading, not compiled, not run on hardware
//
#include <hip/hip_runtime.h>
#include <math.h>

// ---------------------------------------------------------------------------
// EquiPocket forward, MI455X (gfx1250, wave32, WMMA).
// 98% of FLOPs are the per-edge message MLP (262144 pairs x 128x128 GEMM x 4
// layers ~= 34 GFLOP) -> implemented with v_wmma_f32_16x16x32_f16, f32 accum.
// All other stages are tiny (L2-resident) VALU kernels.
// ---------------------------------------------------------------------------

#define NN   512
#define VV   4096
#define HH   128
#define LL   4
#define CUT2 36.0f

typedef __attribute__((ext_vector_type(16))) _Float16 v16h;
typedef __attribute__((ext_vector_type(8)))  _Float16 v8h;
typedef __attribute__((ext_vector_type(8)))  float    v8f;

union U16 { v16h v; v8h h8[2]; };

__device__ __forceinline__ float siluf(float v) { return v / (1.0f + __expf(-v)); }
__device__ __forceinline__ float sigmf_(float v) { return 1.0f / (1.0f + __expf(-v)); }

// ---- pack We2 (per layer) into WMMA B-fragment order: one contiguous v16h per
// lane per (kk,n) fragment.  b[e] = We2[kk*32 + 16*half + e][16*n + lane15].
__global__ void k_pack_we2(const float* __restrict__ We2, _Float16* __restrict__ pack) {
    int idx  = blockIdx.x * 256 + threadIdx.x;      // 65536 total
    int e    = idx & 15;
    int lane = (idx >> 4) & 31;
    int n    = (idx >> 9) & 7;
    int kk   = (idx >> 12) & 3;
    int l    = idx >> 14;
    int hf   = lane >> 4;
    int col  = 16 * n + (lane & 15);
    int K    = kk * 32 + 16 * hf + e;
    pack[idx] = (_Float16)We2[l * 16384 + K * 128 + col];
}

// ---- lg = MLP2(sd, 7->64->64) per vertex
__global__ void k_lg(const float* __restrict__ sd, const float* __restrict__ W1,
                     const float* __restrict__ b1, const float* __restrict__ W2,
                     const float* __restrict__ b2, float* __restrict__ lg) {
    __shared__ float in7[7];
    __shared__ float hid[64];
    int v = blockIdx.x, t = threadIdx.x;            // 64 threads
    if (t < 7) in7[t] = sd[v * 7 + t];
    __syncthreads();
    float s = b1[t];
    for (int k = 0; k < 7; ++k) s += in7[k] * W1[k * 64 + t];
    hid[t] = fmaxf(s, 0.0f);
    __syncthreads();
    float o = b2[t];
    for (int k = 0; k < 64; ++k) o += hid[k] * W2[k * 64 + t];
    lg[v * 64 + t] = o;
}

// ---- segment mean/max over the 8 vertices of each node
__global__ void k_seg(const float* __restrict__ lg, const float* __restrict__ sd,
                      float* __restrict__ geo, float* __restrict__ ss14) {
    int i = blockIdx.x, t = threadIdx.x;            // 64 threads
    float mn = 0.0f, mx = -INFINITY;
    for (int k = 0; k < 8; ++k) {
        float v = lg[(i * 8 + k) * 64 + t];
        mn += v; mx = fmaxf(mx, v);
    }
    geo[i * 128 + t]      = mn * 0.125f;
    geo[i * 128 + 64 + t] = mx;
    if (t < 7) {
        float sm = 0.0f, sx = -INFINITY;
        for (int k = 0; k < 8; ++k) {
            float v = sd[(i * 8 + k) * 7 + t];
            sm += v; sx = fmaxf(sx, v);
        }
        ss14[i * 14 + t]     = sm * 0.125f;
        ss14[i * 14 + 7 + t] = sx;
    }
}

// ---- ss_emb = MLP2(ss14, 14->128->128)
__global__ void k_ss(const float* __restrict__ ss14, const float* __restrict__ W1,
                     const float* __restrict__ b1, const float* __restrict__ W2,
                     const float* __restrict__ b2, float* __restrict__ ssemb) {
    __shared__ float in14[14];
    __shared__ float hid[128];
    int i = blockIdx.x, t = threadIdx.x;            // 128 threads
    if (t < 14) in14[t] = ss14[i * 14 + t];
    __syncthreads();
    float s = b1[t];
    for (int k = 0; k < 14; ++k) s += in14[k] * W1[k * 128 + t];
    hid[t] = fmaxf(s, 0.0f);
    __syncthreads();
    float o = b2[t];
    for (int k = 0; k < 128; ++k) o += hid[k] * W2[k * 128 + t];
    ssemb[i * 128 + t] = o;
}

// ---- h = MLP2(concat(geo, ss_emb), 256->128->128); init x, blocks[0]
__global__ void k_h(const float* __restrict__ geo, const float* __restrict__ ssemb,
                    const float* __restrict__ W1, const float* __restrict__ b1,
                    const float* __restrict__ W2, const float* __restrict__ b2,
                    const float* __restrict__ pos, const float* __restrict__ scp,
                    float* __restrict__ h, float* __restrict__ blocks, float* __restrict__ x) {
    __shared__ float cat[256];
    __shared__ float hid[128];
    int i = blockIdx.x, t = threadIdx.x;            // 128 threads
    cat[t]       = geo[i * 128 + t];
    cat[128 + t] = ssemb[i * 128 + t];
    __syncthreads();
    float s = b1[t];
    for (int k = 0; k < 256; ++k) s += cat[k] * W1[k * 128 + t];
    hid[t] = fmaxf(s, 0.0f);
    __syncthreads();
    float o = b2[t];
    for (int k = 0; k < 128; ++k) o += hid[k] * W2[k * 128 + t];
    h[i * 128 + t]      = o;
    blocks[i * 640 + t] = o;
    if (t < 3)       x[i * 6 + t] = pos[i * 3 + t];
    else if (t < 6)  x[i * 6 + t] = scp[i * 3 + (t - 3)];
}

// ---- deg[i] = max(#neighbors, 1) from original pos
__global__ void k_deg(const float* __restrict__ pos, const int* __restrict__ batch,
                      float* __restrict__ deg) {
    int i = blockIdx.x * blockDim.x + threadIdx.x;
    if (i >= NN) return;
    float px = pos[i * 3], py = pos[i * 3 + 1], pz = pos[i * 3 + 2];
    int b = batch[i];
    float c = 0.0f;
    for (int j = 0; j < NN; ++j) {
        if (j == i || batch[j] != b) continue;
        float dx = px - pos[j * 3], dy = py - pos[j * 3 + 1], dz = pz - pos[j * 3 + 2];
        if (dx * dx + dy * dy + dz * dz < CUT2) c += 1.0f;
    }
    deg[i] = fmaxf(c, 1.0f);
}

// ---- per-layer: hi = h@We_hi[l], hj = h@We_hj[l]; zero agg/xacc
__global__ void k_prep(const float* __restrict__ h, const float* __restrict__ We_hi,
                       const float* __restrict__ We_hj, float* __restrict__ hi,
                       float* __restrict__ hj, float* __restrict__ agg,
                       float* __restrict__ xacc, int l) {
    __shared__ float row[128];
    int i = blockIdx.x, t = threadIdx.x;            // 128 threads
    row[t] = h[i * 128 + t];
    __syncthreads();
    const float* Wi = We_hi + l * 16384;
    const float* Wj = We_hj + l * 16384;
    float si = 0.0f, sj = 0.0f;
    for (int k = 0; k < 128; ++k) {
        float hv = row[k];
        si += hv * Wi[k * 128 + t];
        sj += hv * Wj[k * 128 + t];
    }
    hi[i * 128 + t] = si;
    hj[i * 128 + t] = sj;
    agg[i * 128 + t] = 0.0f;
    if (t < 6) xacc[i * 6 + t] = 0.0f;
}

// ---- THE heavy kernel: per 16x16 node-tile pair, edge-message MLP via WMMA.
//   A (256 pairs x 128, f16 in LDS) = silu(hi_i + hj_j + radial@We_r + be1)
//   m = silu(A @ We2 + be2) * mask ; agg[i] += sum_j m ; w = m@Wx ;
//   xacc[i] += sum_j diff_ij * w.
__global__ __launch_bounds__(256) void k_edge(
    const float* __restrict__ pos, const int* __restrict__ batch,
    const float* __restrict__ x, const float* __restrict__ hi,
    const float* __restrict__ hj, const float* __restrict__ We_r,
    const float* __restrict__ be1, const float* __restrict__ be2,
    const _Float16* __restrict__ We2pack, const float* __restrict__ Wx,
    float* __restrict__ agg, float* __restrict__ xacc, int l) {
    __shared__ alignas(32) _Float16 Alds[256][128];   // 64 KB

    int i0 = blockIdx.y * 16, j0 = blockIdx.x * 16;
    if (batch[i0] != batch[j0]) return;               // uniform: tiles are graph-aligned

    int tid  = threadIdx.x;
    int lane = tid & 31, wave = tid >> 5;
    int hf   = lane >> 4, l15 = lane & 15;

    // per-lane masks for the epilogue rows it will own: pair p = 16*mt + r + 8*hf
    float mreg[2][8];
    for (int mt2 = 0; mt2 < 2; ++mt2) {
        int i = i0 + wave * 2 + mt2;
        float px = pos[i * 3], py = pos[i * 3 + 1], pz = pos[i * 3 + 2];
        for (int r = 0; r < 8; ++r) {
            int j = j0 + r + 8 * hf;
            float dx = px - pos[j * 3], dy = py - pos[j * 3 + 1], dz = pz - pos[j * 3 + 2];
            float d2 = dx * dx + dy * dy + dz * dz;
            mreg[mt2][r] = (i != j && d2 < CUT2) ? 1.0f : 0.0f;
        }
    }

    // Fill A tile: thread tid owns pair p = tid (ii = p>>4, jj = p&15)
    {
        int p = tid, ii = p >> 4, jj = p & 15;
        int i = i0 + ii, j = j0 + jj;
        float r0 = 0.0f, r1 = 0.0f;
        for (int d = 0; d < 3; ++d) { float dd = x[i * 6 + d]     - x[j * 6 + d];     r0 += dd * dd; }
        for (int d = 0; d < 3; ++d) { float dd = x[i * 6 + 3 + d] - x[j * 6 + 3 + d]; r1 += dd * dd; }
        const float* hir = hi  + i * 128;
        const float* hjr = hj  + j * 128;
        const float* w0  = We_r + l * 256;
        const float* w1  = w0 + 128;
        const float* bb  = be1 + l * 128;
        for (int f = 0; f < 128; ++f) {
            float pre = hir[f] + hjr[f] + r0 * w0[f] + r1 * w1[f] + bb[f];
            Alds[p][f] = (_Float16)siluf(pre);
        }
    }
    __syncthreads();

    // GEMM: wave owns M-subtiles {2w, 2w+1} x 8 N-tiles, accumulate 4 K-chunks.
    v8f acc[2][8];
    v8f zero = {};
    for (int a = 0; a < 2; ++a)
        for (int n = 0; n < 8; ++n) acc[a][n] = zero;

    const _Float16* wp = We2pack + l * 16384;
    for (int kk = 0; kk < 4; ++kk) {
        U16 a0, a1;
        int p0 = (wave * 2) * 16 + l15;
        int p1 = (wave * 2 + 1) * 16 + l15;
        int kb = kk * 32 + 16 * hf;
        a0.h8[0] = *(const v8h*)&Alds[p0][kb];
        a0.h8[1] = *(const v8h*)&Alds[p0][kb + 8];
        a1.h8[0] = *(const v8h*)&Alds[p1][kb];
        a1.h8[1] = *(const v8h*)&Alds[p1][kb + 8];
        for (int n = 0; n < 8; ++n) {
            U16 b;
            b.v = *(const v16h*)(wp + kk * 4096 + n * 512 + lane * 16);
            acc[0][n] = __builtin_amdgcn_wmma_f32_16x16x32_f16(
                false, a0.v, false, b.v, (short)0, acc[0][n], false, false);
            acc[1][n] = __builtin_amdgcn_wmma_f32_16x16x32_f16(
                false, a1.v, false, b.v, (short)0, acc[1][n], false, false);
        }
    }

    // Epilogue: m = silu(acc + be2)*mask; agg + coordinate update.
    const float* b2r = be2 + l * 128;
    const float* wxr = Wx  + l * 256;
    for (int mt2 = 0; mt2 < 2; ++mt2) {
        int mt = wave * 2 + mt2;
        int i  = i0 + mt;
        float pw0[8], pw1[8];
        for (int r = 0; r < 8; ++r) { pw0[r] = 0.0f; pw1[r] = 0.0f; }
        for (int n = 0; n < 8; ++n) {
            int col   = 16 * n + l15;
            float bb  = b2r[col];
            float wx0 = wxr[col * 2 + 0];
            float wx1 = wxr[col * 2 + 1];
            float asum = 0.0f;
            v8f c = acc[mt2][n];
            for (int r = 0; r < 8; ++r) {
                float s = siluf(c[r] + bb) * mreg[mt2][r];
                asum   += s;
                pw0[r] += s * wx0;
                pw1[r] += s * wx1;
            }
            asum += __shfl_xor(asum, 16, 32);        // combine the two row-halves
            if (hf == 0) atomicAdd(&agg[i * 128 + col], asum);
        }
        // reduce w = m@Wx across the 16 column-lanes of each half-group
        for (int r = 0; r < 8; ++r) {
            float w0 = pw0[r], w1 = pw1[r];
            for (int d = 1; d < 16; d <<= 1) {
                w0 += __shfl_xor(w0, d, 32);
                w1 += __shfl_xor(w1, d, 32);
            }
            if (l15 == 0 && mreg[mt2][r] != 0.0f) {
                int j = j0 + r + 8 * hf;
                for (int c2 = 0; c2 < 2; ++c2) {
                    float wv = c2 ? w1 : w0;
                    for (int d = 0; d < 3; ++d) {
                        float diff = x[i * 6 + c2 * 3 + d] - x[j * 6 + c2 * 3 + d];
                        atomicAdd(&xacc[i * 6 + c2 * 3 + d], diff * wv);
                    }
                }
            }
        }
    }
}

// ---- per-layer node update: x += xacc/deg ; h += MLP2(concat(h,agg))
__global__ void k_update(float* __restrict__ h, const float* __restrict__ agg,
                         float* __restrict__ x, const float* __restrict__ xacc,
                         const float* __restrict__ deg, const float* __restrict__ Wh1,
                         const float* __restrict__ bh1, const float* __restrict__ Wh2,
                         const float* __restrict__ bh2, float* __restrict__ blocks, int l) {
    __shared__ float cat[256];
    __shared__ float hid[128];
    int i = blockIdx.x, t = threadIdx.x;            // 128 threads
    cat[t]       = h[i * 128 + t];
    cat[128 + t] = agg[i * 128 + t];
    if (t < 6) x[i * 6 + t] += xacc[i * 6 + t] / deg[i];
    __syncthreads();
    const float* W1 = Wh1 + l * 32768;
    const float* W2 = Wh2 + l * 16384;
    float s = bh1[l * 128 + t];
    for (int k = 0; k < 256; ++k) s += cat[k] * W1[k * 128 + t];
    hid[t] = fmaxf(s, 0.0f);
    __syncthreads();
    float o = bh2[l * 128 + t];
    for (int k = 0; k < 128; ++k) o += hid[k] * W2[k * 128 + t];
    float hn = cat[t] + o;
    h[i * 128 + t] = hn;
    blocks[i * 640 + (l + 1) * 128 + t] = hn;
}

// ---- ratios -> att = sigmoid(MLP2) ; scale blocks in place ; angle output
__global__ void k_att(const float* __restrict__ x, const float* __restrict__ pos,
                      const float* __restrict__ W1, const float* __restrict__ b1,
                      const float* __restrict__ W2, const float* __restrict__ b2,
                      float* __restrict__ blocks, float* __restrict__ angle) {
    __shared__ float ratio[6];
    __shared__ float hid[128];
    __shared__ float attL[5];
    int i = blockIdx.x, t = threadIdx.x;            // 128 threads
    if (t == 0) ratio[0] = 1.0f;
    if (t >= 1 && t < 6) {
        float px = x[i * 6], py = x[i * 6 + 1], pz = x[i * 6 + 2];
        float th = (6.0f * t) * (6.0f * t);
        float c = 0.0f;
        for (int j = 0; j < NN; ++j) {
            float dx = px - x[j * 6], dy = py - x[j * 6 + 1], dz = pz - x[j * 6 + 2];
            if (dx * dx + dy * dy + dz * dz < th) c += 1.0f;
        }
        ratio[t] = c / (float)NN;
    }
    __syncthreads();
    float s = b1[t];
    for (int k = 0; k < 6; ++k) s += ratio[k] * W1[k * 128 + t];
    hid[t] = fmaxf(s, 0.0f);
    __syncthreads();
    if (t < 5) {
        float o = b2[t];
        for (int k = 0; k < 128; ++k) o += hid[k] * W2[k * 5 + t];
        attL[t] = sigmf_(o);
    }
    __syncthreads();
    for (int b = 0; b < 5; ++b) blocks[i * 640 + b * 128 + t] *= attL[b];
    if (t < 3) angle[i * 3 + t] = x[i * 6 + t] - pos[i * 3 + t];
}

// ---- y_hat = MLP2(emb, 640->640->1)
__global__ __launch_bounds__(256) void k_out(const float* __restrict__ blocks,
                      const float* __restrict__ W1, const float* __restrict__ b1,
                      const float* __restrict__ W2, const float* __restrict__ b2,
                      float* __restrict__ y) {
    __shared__ float emb[640];
    __shared__ float hid[640];
    __shared__ float red[256];
    int i = blockIdx.x, t = threadIdx.x;            // 256 threads
    for (int k = t; k < 640; k += 256) emb[k] = blocks[i * 640 + k];
    __syncthreads();
    for (int o = t; o < 640; o += 256) {
        float s = b1[o];
        for (int k = 0; k < 640; ++k) s += emb[k] * W1[k * 640 + o];
        hid[o] = fmaxf(s, 0.0f);
    }
    __syncthreads();
    float p = 0.0f;
    for (int k = t; k < 640; k += 256) p += hid[k] * W2[k];
    red[t] = p;
    __syncthreads();
    for (int st = 128; st > 0; st >>= 1) {
        if (t < st) red[t] += red[t + st];
        __syncthreads();
    }
    if (t == 0) y[i] = red[0] + b2[0];
}

extern "C" void kernel_launch(void* const* d_in, const int* in_sizes, int n_in,
                              void* d_out, int out_size, void* d_ws, size_t ws_size,
                              hipStream_t stream) {
    (void)in_sizes; (void)n_in; (void)out_size; (void)ws_size;
    // Inputs, setup_inputs() depth-first insertion order:
    const float* pos   = (const float*)d_in[0];
    const float* scp   = (const float*)d_in[1];
    const float* sd    = (const float*)d_in[2];
    const int*   batch = (const int*)d_in[3];
    /* d_in[4] = vert_batch (implicit: node = vertex/8) */
    const float* lgW1 = (const float*)d_in[5];  const float* lgb1 = (const float*)d_in[6];
    const float* lgW2 = (const float*)d_in[7];  const float* lgb2 = (const float*)d_in[8];
    const float* sfW1 = (const float*)d_in[9];  const float* sfb1 = (const float*)d_in[10];
    const float* sfW2 = (const float*)d_in[11]; const float* sfb2 = (const float*)d_in[12];
    const float* geW1 = (const float*)d_in[13]; const float* geb1 = (const float*)d_in[14];
    const float* geW2 = (const float*)d_in[15]; const float* geb2 = (const float*)d_in[16];
    const float* atW1 = (const float*)d_in[17]; const float* atb1 = (const float*)d_in[18];
    const float* atW2 = (const float*)d_in[19]; const float* atb2 = (const float*)d_in[20];
    const float* ouW1 = (const float*)d_in[21]; const float* oub1 = (const float*)d_in[22];
    const float* ouW2 = (const float*)d_in[23]; const float* oub2 = (const float*)d_in[24];
    const float* We_hi = (const float*)d_in[25];
    const float* We_hj = (const float*)d_in[26];
    const float* We_r  = (const float*)d_in[27];
    const float* be1   = (const float*)d_in[28];
    const float* We2   = (const float*)d_in[29];
    const float* be2   = (const float*)d_in[30];
    const float* Wx    = (const float*)d_in[31];
    const float* Wh1   = (const float*)d_in[32];
    const float* bh1   = (const float*)d_in[33];
    const float* Wh2   = (const float*)d_in[34];
    const float* bh2   = (const float*)d_in[35];

    // Workspace layout (floats), ~4.1 MB total.
    float* ws     = (float*)d_ws;
    float* lg     = ws;                    // 262144
    float* geo    = ws + 262144;           // 65536
    float* ss14   = ws + 327680;           // 7168
    float* ssemb  = ws + 334848;           // 65536
    float* h      = ws + 400384;           // 65536
    float* x      = ws + 465920;           // 3072
    float* deg    = ws + 468992;           // 512
    float* agg    = ws + 469504;           // 65536
    float* xacc   = ws + 535040;           // 3072
    float* blocks = ws + 538112;           // 327680
    float* hib    = ws + 865792;           // 65536
    float* hjb    = ws + 931328;           // 65536
    _Float16* We2pack = (_Float16*)(ws + 996864);   // 65536 halves (32B aligned)

    float* y   = (float*)d_out;            // [0,512)   y_hat
    float* ang = y + NN;                   // [512,2048) angle

    k_pack_we2<<<256, 256, 0, stream>>>(We2, We2pack);
    k_lg<<<VV, 64, 0, stream>>>(sd, lgW1, lgb1, lgW2, lgb2, lg);
    k_seg<<<NN, 64, 0, stream>>>(lg, sd, geo, ss14);
    k_ss<<<NN, 128, 0, stream>>>(ss14, sfW1, sfb1, sfW2, sfb2, ssemb);
    k_h<<<NN, 128, 0, stream>>>(geo, ssemb, geW1, geb1, geW2, geb2, pos, scp, h, blocks, x);
    k_deg<<<4, 128, 0, stream>>>(pos, batch, deg);

    for (int l = 0; l < LL; ++l) {
        k_prep<<<NN, 128, 0, stream>>>(h, We_hi, We_hj, hib, hjb, agg, xacc, l);
        dim3 g(32, 32);
        k_edge<<<g, 256, 0, stream>>>(pos, batch, x, hib, hjb, We_r, be1, be2,
                                      We2pack, Wx, agg, xacc, l);
        k_update<<<NN, 128, 0, stream>>>(h, agg, x, xacc, deg, Wh1, bh1, Wh2, bh2, blocks, l);
    }

    k_att<<<NN, 128, 0, stream>>>(x, pos, atW1, atb1, atW2, atb2, blocks, ang);
    k_out<<<NN, 256, 0, stream>>>(blocks, ouW1, oub1, ouW2, oub2, y);
}
